// RelationalGATLayer_2628519985723
// MI455X (gfx1250) — compile-verified, matrix operand order checked by reading, not run on hardware
//
#include <hip/hip_runtime.h>

typedef __attribute__((ext_vector_type(16))) _Float16 v16h;
typedef __attribute__((ext_vector_type(8)))  float    v8f;

#define IN_DIM   128
#define HC       256   // HEADS * OUT_DIM
#define OUT_DIM  64
#define HEADS    4
#define NEG_SLOPE 0.2f

// ---------------------------------------------------------------------------
// Kernel 1: x_proj = x @ W^T via V_WMMA_F32_16X16X32_F16.
// One wave -> one 16x16 output tile, K=128 as 4 WMMA steps of K=32.
// Block = 512 threads = 16 waves -> 16 rows x 256 cols per block.
// ---------------------------------------------------------------------------
__global__ __launch_bounds__(512) void rgat_proj_wmma(
    const float* __restrict__ x, const float* __restrict__ W,
    float* __restrict__ xproj, int n) {
  const int lane  = threadIdx.x & 31;
  const int wave  = threadIdx.x >> 5;          // 0..15 -> column tile
  const int row0  = blockIdx.x * 16;
  const int col0  = wave * 16;
  const int r     = lane & 15;                 // A row / B col within tile
  const int khalf = (lane >> 4) ? 8 : 0;       // K sub-pattern per lane half

  int arow = row0 + r;
  if (arow >= n) arow = n - 1;                 // safe clamp (n%16==0 normally)

  v8f acc = {};
  const float* aprow = x + (size_t)arow * IN_DIM + khalf;
  const float* bprow = W + (size_t)(col0 + r) * IN_DIM + khalf;

  #pragma unroll
  for (int k0 = 0; k0 < IN_DIM; k0 += 32) {
    v16h a, b;
    #pragma unroll
    for (int i = 0; i < 8; ++i) {
      // VGPR i holds K pair {kk, kk+1}; i<4 -> K base 0, i>=4 -> K base 16
      const int kk = k0 + (i & 3) * 2 + (i >> 2) * 16;
      a[2 * i]     = (_Float16)aprow[kk];
      a[2 * i + 1] = (_Float16)aprow[kk + 1];
      b[2 * i]     = (_Float16)bprow[kk];
      b[2 * i + 1] = (_Float16)bprow[kk + 1];
    }
    acc = __builtin_amdgcn_wmma_f32_16x16x32_f16(
        /*neg_a=*/false, a, /*neg_b=*/false, b,
        /*c_mod=*/(short)0, acc, /*reuse_a=*/false, /*reuse_b=*/false);
  }

  // D layout: VGPR i -> M = i + (lane>=16 ? 8 : 0), N = lane&15
  const int mbase = row0 + ((lane >> 4) << 3);
  const int col   = col0 + (lane & 15);
  #pragma unroll
  for (int i = 0; i < 8; ++i) {
    const int rr = mbase + i;
    if (rr < n) xproj[(size_t)rr * HC + col] = acc[i];
  }
}

// ---------------------------------------------------------------------------
// Kernel 2: per-node attention logits a_src/a_dst [N,H]
// ---------------------------------------------------------------------------
__global__ void rgat_node_att(const float* __restrict__ xproj,
                              const float* __restrict__ att_src,
                              const float* __restrict__ att_dst,
                              float* __restrict__ a_src,
                              float* __restrict__ a_dst, int n) {
  const int idx = blockIdx.x * blockDim.x + threadIdx.x;   // node*H + h
  if (idx >= n * HEADS) return;
  const int node = idx >> 2, h = idx & 3;
  const float* row = xproj + (size_t)node * HC + h * OUT_DIM;
  float s = 0.f, d = 0.f;
  #pragma unroll 8
  for (int c = 0; c < OUT_DIM; ++c) {
    const float v = row[c];
    s += v * att_src[h * OUT_DIM + c];
    d += v * att_dst[h * OUT_DIM + c];
  }
  a_src[idx] = s;
  a_dst[idx] = d;
}

// ---------------------------------------------------------------------------
// Kernel 3: per-relation logits a_rel [R,H]  (tiny: 64x4, one block)
// ---------------------------------------------------------------------------
__global__ void rgat_rel_att(const float* __restrict__ rel_emb,
                             const float* __restrict__ W_edge,
                             const float* __restrict__ att_edge,
                             float* __restrict__ a_rel, int nrel) {
  const int idx = blockIdx.x * blockDim.x + threadIdx.x;
  if (idx >= nrel * HEADS) return;
  const int rIdx = idx >> 2, h = idx & 3;
  const float* er = rel_emb + (size_t)rIdx * IN_DIM;
  float acc = 0.f;
  for (int c = 0; c < OUT_DIM; ++c) {
    const float* wrow = W_edge + (size_t)(h * OUT_DIM + c) * IN_DIM;
    float p = 0.f;
    #pragma unroll 8
    for (int k = 0; k < IN_DIM; ++k) p += er[k] * wrow[k];
    acc += p * att_edge[h * OUT_DIM + c];
  }
  a_rel[idx] = acc;
}

// ---------------------------------------------------------------------------
// Init kernels
// ---------------------------------------------------------------------------
__global__ void rgat_init_mz(int* __restrict__ m_bits, float* __restrict__ z, int nh) {
  const int i = blockIdx.x * blockDim.x + threadIdx.x;
  if (i < nh) { m_bits[i] = (int)0xFF800000; z[i] = 0.f; }   // -inf, 0
}

__global__ void rgat_init_out(float* __restrict__ out, const float* __restrict__ bias, int tot) {
  const int i = blockIdx.x * blockDim.x + threadIdx.x;
  if (i < tot) out[i] = bias[i & (OUT_DIM - 1)];
}

// ---------------------------------------------------------------------------
// Kernel 4: per-edge logit + leaky-relu + atomic segment-max (bit trick)
// ---------------------------------------------------------------------------
__global__ void rgat_edge_alpha_max(const long long* __restrict__ ei,
                                    const long long* __restrict__ et,
                                    const float* __restrict__ a_src,
                                    const float* __restrict__ a_dst,
                                    const float* __restrict__ a_rel,
                                    float* __restrict__ alpha_buf,
                                    int* __restrict__ m_bits, int ne) {
  const int e = blockIdx.x * blockDim.x + threadIdx.x;
  if (e >= ne) return;
  const int s = (int)ei[e], d = (int)ei[ne + e], t = (int)et[e];
  #pragma unroll
  for (int h = 0; h < HEADS; ++h) {
    float a = a_src[s * HEADS + h] + a_dst[d * HEADS + h] + a_rel[t * HEADS + h];
    a = a > 0.f ? a : NEG_SLOPE * a;
    alpha_buf[(size_t)e * HEADS + h] = a;
    int* addr = m_bits + d * HEADS + h;
    const int bits = __float_as_int(a);
    if (bits >= 0) atomicMax(addr, bits);
    else           atomicMin((unsigned int*)addr, (unsigned int)bits);
  }
}

// segment_max of empty segment -> -inf -> reference maps non-finite to 0
__global__ void rgat_fix_m(int* __restrict__ m_bits, float* __restrict__ m, int nh) {
  const int i = blockIdx.x * blockDim.x + threadIdx.x;
  if (i >= nh) return;
  const int bits = m_bits[i];
  const bool fin = (bits & 0x7F800000) != 0x7F800000;
  m[i] = fin ? __int_as_float(bits) : 0.f;
}

// ---------------------------------------------------------------------------
// Kernel 5: ea = exp(alpha - m[dst]); z[dst] += ea (atomic); alpha_buf <- ea
// ---------------------------------------------------------------------------
__global__ void rgat_edge_exp_sum(const long long* __restrict__ ei,
                                  float* __restrict__ alpha_buf,
                                  const float* __restrict__ m,
                                  float* __restrict__ z, int ne) {
  const int e = blockIdx.x * blockDim.x + threadIdx.x;
  if (e >= ne) return;
  const int d = (int)ei[ne + e];
  #pragma unroll
  for (int h = 0; h < HEADS; ++h) {
    const float ea = __expf(alpha_buf[(size_t)e * HEADS + h] - m[d * HEADS + h]);
    alpha_buf[(size_t)e * HEADS + h] = ea;
    atomicAdd(&z[d * HEADS + h], ea);
  }
}

// ---------------------------------------------------------------------------
// Kernel 6: weighted message scatter with head-mean folded in.
// 64 threads per edge (one per output channel); head-mean before the atomic
// -> 64 atomics/edge instead of 256.
// ---------------------------------------------------------------------------
__global__ __launch_bounds__(256) void rgat_edge_scatter(
    const long long* __restrict__ ei, const float* __restrict__ alpha_buf,
    const float* __restrict__ z, const float* __restrict__ xproj,
    float* __restrict__ out, int ne) {
  const int gid = blockIdx.x * blockDim.x + threadIdx.x;
  const int e = gid >> 6, c = gid & 63;
  if (e >= ne) return;
  const int s = (int)ei[e], d = (int)ei[ne + e];
  const float* xr = xproj + (size_t)s * HC + c;
  float acc = 0.f;
  #pragma unroll
  for (int h = 0; h < HEADS; ++h) {
    const float w = alpha_buf[(size_t)e * HEADS + h] /
                    (z[d * HEADS + h] + 1e-16f);
    acc += w * xr[h * OUT_DIM];
  }
  atomicAdd(&out[(size_t)d * OUT_DIM + c], 0.25f * acc);
}

// ---------------------------------------------------------------------------
// Launch
// ---------------------------------------------------------------------------
extern "C" void kernel_launch(void* const* d_in, const int* in_sizes, int n_in,
                              void* d_out, int out_size, void* d_ws, size_t ws_size,
                              hipStream_t stream) {
  const float*     x        = (const float*)d_in[0];
  const long long* ei       = (const long long*)d_in[1];   // int64 [2,E]
  const long long* et       = (const long long*)d_in[2];   // int64 [E]
  const float*     rel_emb  = (const float*)d_in[3];
  const float*     W        = (const float*)d_in[4];
  const float*     att_src  = (const float*)d_in[5];
  const float*     att_dst  = (const float*)d_in[6];
  const float*     W_edge   = (const float*)d_in[7];
  const float*     att_edge = (const float*)d_in[8];
  const float*     bias     = (const float*)d_in[9];
  float*           out      = (float*)d_out;

  const int n    = in_sizes[0] / IN_DIM;   // 50000
  const int ne   = in_sizes[1] / 2;        // 800000
  const int nrel = in_sizes[3] / IN_DIM;   // 64

  // Workspace layout (256B aligned slices)
  char* ws = (char*)d_ws;
  size_t off = 0;
  auto take = [&](size_t bytes) -> char* {
    char* p = ws + off;
    off = (off + bytes + 255) & ~(size_t)255;
    return p;
  };
  float* xproj     = (float*)take((size_t)n * HC * sizeof(float));
  float* a_src     = (float*)take((size_t)n * HEADS * sizeof(float));
  float* a_dst     = (float*)take((size_t)n * HEADS * sizeof(float));
  float* a_rel     = (float*)take((size_t)nrel * HEADS * sizeof(float));
  int*   m_bits    = (int*)  take((size_t)n * HEADS * sizeof(float));
  float* m         = (float*)m_bits;                         // reused in place
  float* z         = (float*)take((size_t)n * HEADS * sizeof(float));
  float* alpha_buf = (float*)take((size_t)ne * HEADS * sizeof(float));
  (void)ws_size; (void)n_in; (void)out_size;

  const int nh = n * HEADS;

  // Independent setup
  rgat_init_mz<<<(nh + 255) / 256, 256, 0, stream>>>(m_bits, z, nh);
  rgat_init_out<<<(n * OUT_DIM + 255) / 256, 256, 0, stream>>>(out, bias, n * OUT_DIM);
  rgat_rel_att<<<(nrel * HEADS + 255) / 256, 256, 0, stream>>>(rel_emb, W_edge, att_edge, a_rel, nrel);

  // Dense projection (WMMA)
  rgat_proj_wmma<<<(n + 15) / 16, 512, 0, stream>>>(x, W, xproj, n);

  // Node logits
  rgat_node_att<<<(nh + 255) / 256, 256, 0, stream>>>(xproj, att_src, att_dst, a_src, a_dst, n);

  // Edge phase: max -> exp/sum -> normalize+scatter
  rgat_edge_alpha_max<<<(ne + 255) / 256, 256, 0, stream>>>(ei, et, a_src, a_dst, a_rel,
                                                            alpha_buf, m_bits, ne);
  rgat_fix_m<<<(nh + 255) / 256, 256, 0, stream>>>(m_bits, m, nh);
  rgat_edge_exp_sum<<<(ne + 255) / 256, 256, 0, stream>>>(ei, alpha_buf, m, z, ne);
  rgat_edge_scatter<<<((size_t)ne * 64 + 255) / 256, 256, 0, stream>>>(ei, alpha_buf, z,
                                                                       xproj, out, ne);
}
